// EMAKMeansVectorQuantizer_52123723105004
// MI455X (gfx1250) — matrix operand order repaired and checked by reading, not compiled, not run on hardware
//
#include <hip/hip_runtime.h>
#include <hip/hip_bf16.h>

#define BATCH 32
#define DIM   256
#define TLEN  4096
#define KCODES 512
#define NROWS  (BATCH * TLEN)          // 131072 query vectors
#define ROWS_PER_WG 16
#define NWG (NROWS / ROWS_PER_WG)      // 8192 workgroups

typedef __attribute__((ext_vector_type(16))) __bf16 v16bf;
typedef __attribute__((ext_vector_type(8)))  float  v8f;

union Frag { unsigned int u[8]; v16bf v; uint4 q[2]; };

__device__ __forceinline__ unsigned short f32_bf16_rne(float f) {
    unsigned int u = __float_as_uint(f);
    u += 0x7FFFu + ((u >> 16) & 1u);   // round-to-nearest-even
    return (unsigned short)(u >> 16);
}

// ---------------------------------------------------------------------------
// Kernel 1: codebook prep. embedBF = bf16(-2*e) (the -2 folded into B matrix),
// enorm[k] = |e_k|^2 (becomes the WMMA C-init), and zero the histogram.
// ---------------------------------------------------------------------------
__global__ __launch_bounds__(256) void vq_prep(const float* __restrict__ embed,
                                               unsigned short* __restrict__ embedBF,
                                               float* __restrict__ enorm,
                                               unsigned int* __restrict__ counts) {
    __shared__ float red[256];
    const int k = blockIdx.x, d = threadIdx.x;
    float v = embed[k * DIM + d];
    embedBF[k * DIM + d] = f32_bf16_rne(-2.0f * v);
    red[d] = v * v;
    __syncthreads();
    for (int s = 128; s > 0; s >>= 1) { if (d < s) red[d] += red[d + s]; __syncthreads(); }
    if (d == 0) { enorm[k] = red[0]; counts[k] = 0u; }
}

// ---------------------------------------------------------------------------
// Kernel 2: main VQ kernel. One block = 16 rows (never crosses a batch item,
// T%16==0). 8 waves x 4 col-tiles cover all 512 codes. Score tile computed as
// C = |e|^2 + (x_bf16) . (-2 e_bf16)  via 8 chained v_wmma_f32_16x16x32_bf16.
// ---------------------------------------------------------------------------
__global__ __launch_bounds__(256) void vq_main(const float* __restrict__ inputs,
                                               const float* __restrict__ embed,
                                               const unsigned short* __restrict__ embedBF,
                                               const float* __restrict__ enorm,
                                               unsigned int* __restrict__ counts,
                                               float* __restrict__ partials,
                                               float* __restrict__ zq) {
    __shared__ unsigned short xbf[ROWS_PER_WG * 264]; // bf16 x tile, pad 8 -> conflict-free A-frag reads
    __shared__ float xf[ROWS_PER_WG * 257];           // f32 x tile for the loss, pad 1
    __shared__ float scores[ROWS_PER_WG * 516];       // 16 x 512 scores, pad 4
    __shared__ int   qidx[ROWS_PER_WG];
    __shared__ float red[256];

    const int tid  = threadIdx.x;
    const int g    = blockIdx.x;
    const int row0 = g * ROWS_PER_WG;
    const int b    = row0 / TLEN;
    const int t0   = row0 % TLEN;

    // ---- Phase 1: stage x tile (coalesced 64B segments along t) ----
    {
        const int tsub = tid & 15, dgrp = tid >> 4;
        #pragma unroll
        for (int dd = 0; dd < 16; ++dd) {
            int d = dgrp * 16 + dd;
            float v = inputs[(b * DIM + d) * TLEN + t0 + tsub];
            xf [tsub * 257 + d] = v;
            xbf[tsub * 264 + d] = f32_bf16_rne(v);
        }
    }
    __syncthreads();

    const int lane = tid & 31, wid = tid >> 5;
    const int m = lane & 15, kh = lane >> 4;   // A row / K-half per ISA 16-bit A layout

    // ---- Hoist all 8 A fragments (whole D=256 slice) into registers ----
    // 16-bit A 16x32 layout: lane m holds K in {kh*8+0..7} U {16+kh*8+0..7},
    // VGPR j packs (K=2j, 2j+1) within each group of 8.
    Frag afr[8];
    #pragma unroll
    for (int kk = 0; kk < 8; ++kk) {
        #pragma unroll
        for (int vg = 0; vg < 8; ++vg) {
            int kb = (vg < 4) ? (kh * 8 + 2 * vg) : (16 + kh * 8 + 2 * (vg - 4));
            int d  = kk * 32 + kb;
            afr[kk].u[vg] = ((const unsigned int*)xbf)[(m * 264 + d) >> 1];
        }
    }

    // ---- Phase 2: 4 col-tiles per wave, 8 WMMAs each over D=256 ----
    #pragma unroll
    for (int ctl = 0; ctl < 4; ++ctl) {
        const int code = (wid * 4 + ctl) * 16 + m;   // B-matrix n = lane&15
        const float en = enorm[code];
        v8f c = { en, en, en, en, en, en, en, en };  // C-init = |e|^2
        // B 32x16 layout (matches A): lane half kh holds K in {kh*8+0..7} U {16+kh*8+0..7};
        // per lane two runs of 8 bf16 of the (-2e) row -> two 16B loads.
        const unsigned short* bp = embedBF + code * DIM + kh * 8;
        #pragma unroll
        for (int kk = 0; kk < 8; ++kk) {
            Frag bfr;
            const uint4* q = (const uint4*)(bp + kk * 32);
            bfr.q[0] = q[0];
            bfr.q[1] = q[2];
            c = __builtin_amdgcn_wmma_f32_16x16x32_bf16(
                    false, afr[kk].v, false, bfr.v, (short)0, c, false, false);
        }
        // C layout: VGPR r -> row (kh*8 + r), column = code
        #pragma unroll
        for (int r = 0; r < 8; ++r)
            scores[(kh * 8 + r) * 516 + code] = c[r];
    }
    __syncthreads();

    // ---- Phase 3: argmin over 512 codes (16 threads/row, wave32 shuffles) ----
    {
        const int row = tid >> 4, sub = tid & 15;
        float best = 3.402823466e38f; int bidx = 0;
        #pragma unroll
        for (int j = 0; j < 32; ++j) {
            int code = j * 16 + sub;               // ascending -> first-min per thread
            float s = scores[row * 516 + code];
            if (s < best) { best = s; bidx = code; }
        }
        #pragma unroll
        for (int off = 1; off <= 8; off <<= 1) {   // stays within 16-lane halves
            float ob = __shfl_xor(best, off, 32);
            int   oi = __shfl_xor(bidx, off, 32);
            if (ob < best || (ob == best && oi < bidx)) { best = ob; bidx = oi; }
        }
        if (sub == 0) {
            qidx[row] = bidx;
            atomicAdd(&counts[bidx], 1u);          // deterministic integer histogram
        }
    }
    __syncthreads();

    // ---- Phase 4: gather z_q (coalesced along t), accumulate loss partial ----
    float sumsq = 0.0f;
    {
        const int tsub = tid & 15, dgrp = tid >> 4;
        const float* ev = embed + qidx[tsub] * DIM;  // codebook is L2-hot
        #pragma unroll
        for (int dd = 0; dd < 16; ++dd) {
            int d = dgrp * 16 + dd;
            float val  = ev[d];
            float diff = val - xf[tsub * 257 + d];
            sumsq += diff * diff;
            zq[(b * DIM + d) * TLEN + t0 + tsub] = val;
        }
    }
    red[tid] = sumsq;
    __syncthreads();
    for (int s = 128; s > 0; s >>= 1) { if (tid < s) red[tid] += red[tid + s]; __syncthreads(); }
    if (tid == 0) partials[g] = red[0];
}

// ---------------------------------------------------------------------------
// Kernel 3: scalars. loss = 0.25 * sum / (N*D); perplexity from histogram;
// kldiv_r = log(512) * T broadcast over B.
// ---------------------------------------------------------------------------
__global__ __launch_bounds__(256) void vq_final(const float* __restrict__ partials,
                                                const unsigned int* __restrict__ counts,
                                                float* __restrict__ outScalars) {
    __shared__ float red[256];
    const int tid = threadIdx.x;
    float s = 0.0f;
    for (int i = tid; i < NWG; i += 256) s += partials[i];
    red[tid] = s;
    __syncthreads();
    for (int st = 128; st > 0; st >>= 1) { if (tid < st) red[tid] += red[tid + st]; __syncthreads(); }
    const float lossSum = red[0];
    __syncthreads();

    float h = 0.0f;
    for (int c = tid; c < KCODES; c += 256) {
        float p = (float)counts[c] * (1.0f / (float)NROWS);
        h += p * logf(p + 1e-10f);
    }
    red[tid] = h;
    __syncthreads();
    for (int st = 128; st > 0; st >>= 1) { if (tid < st) red[tid] += red[tid + st]; __syncthreads(); }

    if (tid == 0) {
        outScalars[0]  = 0.25f * lossSum * (1.0f / (float)(NROWS * DIM)); // loss
        outScalars[33] = expf(-red[0]);                                   // perplexity
    }
    if (tid < BATCH)
        outScalars[1 + tid] = 9.0f * 0.69314718055994531f * (float)TLEN;  // log(512)*4096
}

// ---------------------------------------------------------------------------
extern "C" void kernel_launch(void* const* d_in, const int* in_sizes, int n_in,
                              void* d_out, int out_size, void* d_ws, size_t ws_size,
                              hipStream_t stream) {
    (void)in_sizes; (void)n_in; (void)out_size; (void)ws_size;
    const float* inputs = (const float*)d_in[0];   // [B, D, T] f32
    const float* embed  = (const float*)d_in[1];   // [K, D]    f32
    float* out = (float*)d_out;

    char* ws = (char*)d_ws;
    unsigned short* embedBF = (unsigned short*)(ws);                    // 262144 B
    float*        enorm    = (float*)(ws + 262144);                     //   2048 B
    unsigned int* counts   = (unsigned int*)(ws + 262144 + 2048);       //   2048 B
    float*        partials = (float*)(ws + 262144 + 4096);              //  32768 B

    float* zq      = out;                                   // B*D*T floats
    float* scalars = out + (size_t)BATCH * DIM * TLEN;      // loss, kldiv[32], perplexity

    vq_prep <<<KCODES, 256, 0, stream>>>(embed, embedBF, enorm, counts);
    vq_main <<<NWG,    256, 0, stream>>>(inputs, embed, embedBF, enorm, counts, partials, zq);
    vq_final<<<1,      256, 0, stream>>>(partials, counts, scalars);
}
